// CrystalGraphDiffusionModel_35837207117861
// MI455X (gfx1250) — matrix operand order repaired and checked by reading, not compiled, not run on hardware
//
#include <hip/hip_runtime.h>
#include <hip/hip_bf16.h>
#include <math.h>

#define N_NODES 30000
#define N_EDGES 480000
#define N_GRAPHS 16
#define HID 128
#define HEADS 8
#define NODE_IN 64
#define EDGE_IN 32
#define COND 8
#define NODE_OUT 64
#define N_LAYERS 4
#define EF (N_EDGES + N_NODES)

typedef __attribute__((ext_vector_type(16))) _Float16 v16h;
typedef __attribute__((ext_vector_type(8)))  _Float16 v8h;
typedef __attribute__((ext_vector_type(8)))  float    v8f;

__device__ __forceinline__ float gelu_exact(float x) {
  return 0.5f * x * (1.0f + erff(x * 0.70710678118654752440f));
}

// order-preserving float <-> uint mapping for atomic segment-max
__device__ __forceinline__ unsigned fkey(float x) {
  unsigned b = __float_as_uint(x);
  return (b & 0x80000000u) ? ~b : (b | 0x80000000u);
}
__device__ __forceinline__ float funkey(unsigned u) {
  unsigned b = (u & 0x80000000u) ? (u & 0x7FFFFFFFu) : ~u;
  return __uint_as_float(b);
}

// ---------- WMMA fragment helpers (wave32, v_wmma_f32_16x16x32_f16) ----------
// A fragment (16x32 f16) built from f32 row-major source; rows/kb are 32B aligned
__device__ __forceinline__ v16h load_a_f32(const float* __restrict__ A, int lda,
                                           int r0, int k0, int lane) {
  int m = lane & 15;
  int kb = (lane >> 4) * 8;
  const float* row = A + (size_t)(r0 + m) * lda + k0 + kb;
  float4 p0 = *(const float4*)(row);
  float4 p1 = *(const float4*)(row + 4);
  float4 p2 = *(const float4*)(row + 16);
  float4 p3 = *(const float4*)(row + 20);
  v16h a;
  a[0]  = (_Float16)p0.x; a[1]  = (_Float16)p0.y; a[2]  = (_Float16)p0.z; a[3]  = (_Float16)p0.w;
  a[4]  = (_Float16)p1.x; a[5]  = (_Float16)p1.y; a[6]  = (_Float16)p1.z; a[7]  = (_Float16)p1.w;
  a[8]  = (_Float16)p2.x; a[9]  = (_Float16)p2.y; a[10] = (_Float16)p2.z; a[11] = (_Float16)p2.w;
  a[12] = (_Float16)p3.x; a[13] = (_Float16)p3.y; a[14] = (_Float16)p3.z; a[15] = (_Float16)p3.w;
  return a;
}
// A fragment from f16 row-major source (16B-aligned chunks)
__device__ __forceinline__ v16h load_a_f16(const _Float16* __restrict__ A, int lda,
                                           int r0, int k0, int lane) {
  int m = lane & 15;
  int kb = (lane >> 4) * 8;
  const _Float16* row = A + (size_t)(r0 + m) * lda + k0 + kb;
  v8h lo = *(const v8h*)(row);
  v8h hi = *(const v8h*)(row + 16);
  v16h a;
#pragma unroll
  for (int e = 0; e < 8; ++e) { a[e] = lo[e]; a[8 + e] = hi[e]; }
  return a;
}
// B fragment from pre-packed layout: tile (kt*CT + j) holds 32 lanes x 16 halves
__device__ __forceinline__ v16h load_b_packed(const _Float16* __restrict__ Bp,
                                              int CT_, int kt, int j, int lane) {
  return *(const v16h*)(Bp + ((size_t)((kt * CT_ + j) * 32 + lane) << 4));
}
__device__ __forceinline__ v8f wmma_fma(v16h a, v16h b, v8f c) {
  return __builtin_amdgcn_wmma_f32_16x16x32_f16(false, a, false, b, (short)0, c, false, false);
}

// ---------------------------- utility kernels --------------------------------
__global__ void zero_kernel(float* __restrict__ p, long n) {
  long i = (long)blockIdx.x * blockDim.x + threadIdx.x;
  if (i < n) p[i] = 0.0f;
}

// convert f32 weight [K,Ncols] -> f16 packed WMMA-B layout [Kpad/32][Npad/16][32][16]
__global__ void cvt_pack_kernel(const float* __restrict__ src, int K, int Ncols,
                                _Float16* __restrict__ dst, int Kpad, int Npad) {
  int tid = blockIdx.x * blockDim.x + threadIdx.x;
  if (tid >= Kpad * Npad) return;
  int CT_ = Npad >> 4;
  int e = tid & 15;
  int lane = (tid >> 4) & 31;
  int tile = tid >> 9;                 // kt*CT_ + j
  int kt = tile / CT_, j = tile - kt * CT_;
  int k = kt * 32 + (lane >> 4) * 8 + (e & 7) + (e >> 3) * 16;
  int n = j * 16 + (lane & 15);
  float v = (k < K && n < Ncols) ? src[k * Ncols + n] : 0.0f;
  dst[tid] = (_Float16)v;
}

__global__ void time_mlp_kernel(const float* __restrict__ t,
                                const float* __restrict__ tW1, const float* __restrict__ tb1,
                                const float* __restrict__ tW2, const float* __restrict__ tb2,
                                float* __restrict__ temb) {
  __shared__ float hm[HID];
  int b = blockIdx.x;
  int j = threadIdx.x;
  hm[j] = gelu_exact(t[b] * tW1[j] + tb1[j]);
  __syncthreads();
  float s = tb2[j];
  for (int k = 0; k < HID; ++k) s += hm[k] * tW2[k * HID + j];
  temb[b * HID + j] = s;
}

__global__ void concat_kernel(const float* __restrict__ x, const float* __restrict__ cond,
                              const int* __restrict__ batch, float* __restrict__ Acat) {
  int tid = blockIdx.x * blockDim.x + threadIdx.x;
  if (tid >= N_NODES * 96) return;
  int n = tid / 96, c = tid % 96;
  float v = 0.0f;
  if (c < NODE_IN) v = x[n * NODE_IN + c];
  else if (c < NODE_IN + COND) v = cond[batch[n] * COND + (c - NODE_IN)];
  Acat[tid] = v;
}

// ------------------------- generic WMMA GEMM ---------------------------------
// C[M,Ncols] = A[M,Kpad](f32) @ Bpacked[Kpad, CT*16](f16) + bias; one wave = 16 rows
template <int CT>
__global__ void gemm_wmma_kernel(const float* __restrict__ A, int M, int Kpad,
                                 const _Float16* __restrict__ B,
                                 const float* __restrict__ bias,
                                 float* __restrict__ C, int Ncols) {
  int wave = blockIdx.x * (blockDim.x >> 5) + (threadIdx.x >> 5);
  int r0 = wave * 16;
  if (r0 >= M) return;
  int lane = threadIdx.x & 31;
  v8f acc[CT] = {};
  int kts = Kpad >> 5;
  for (int kt = 0; kt < kts; ++kt) {
    v16h a = load_a_f32(A, Kpad, r0, kt * 32, lane);
#pragma unroll
    for (int j = 0; j < CT; ++j) {
      v16h b = load_b_packed(B, CT, kt, j, lane);
      acc[j] = wmma_fma(a, b, acc[j]);
    }
  }
  int mb = (lane >> 4) * 8, nloc = lane & 15;
#pragma unroll
  for (int j = 0; j < CT; ++j) {
    int col = j * 16 + nloc;
    if (col < Ncols) {
      float bv = bias ? bias[col] : 0.0f;
#pragma unroll
      for (int v = 0; v < 8; ++v)
        C[(size_t)(r0 + mb + v) * Ncols + col] = acc[j][v] + bv;
    }
  }
}

// ------------------- LayerNorm + GELU (+optional add/copy) -------------------
__global__ void ln_gelu_kernel(const float* __restrict__ src,
                               const float* __restrict__ g, const float* __restrict__ bb,
                               float* __restrict__ dst, float* __restrict__ dst2,
                               const float* __restrict__ addv, const int* __restrict__ batch,
                               int M) {
  int wave = blockIdx.x * (blockDim.x >> 5) + (threadIdx.x >> 5);
  if (wave >= M) return;
  int lane = threadIdx.x & 31;
  const float* row = src + (size_t)wave * HID;
  float v[4];
  float s = 0.0f;
#pragma unroll
  for (int i = 0; i < 4; ++i) { v[i] = row[lane + 32 * i]; s += v[i]; }
#pragma unroll
  for (int off = 16; off > 0; off >>= 1) s += __shfl_xor(s, off, 32);
  float mean = s * (1.0f / HID);
  float q = 0.0f;
#pragma unroll
  for (int i = 0; i < 4; ++i) { float d = v[i] - mean; q += d * d; }
#pragma unroll
  for (int off = 16; off > 0; off >>= 1) q += __shfl_xor(q, off, 32);
  float rstd = rsqrtf(q * (1.0f / HID) + 1e-5f);
  size_t base = (size_t)wave * HID;
  int bofs = addv ? batch[wave] * HID : 0;
#pragma unroll
  for (int i = 0; i < 4; ++i) {
    int c = lane + 32 * i;
    float y = gelu_exact((v[i] - mean) * rstd * g[c] + bb[c]);
    if (addv) y += addv[bofs + c];
    dst[base + c] = y;
    if (dst2) dst2[base + c] = y;
  }
}

// --------- fused edge encoder: GEMM(K=32) + LN + GELU + loop-mean accum ------
__global__ void edge_encode_kernel(const float* __restrict__ ea_in,
                                   const _Float16* __restrict__ Bw,
                                   const float* __restrict__ eb,
                                   const float* __restrict__ g, const float* __restrict__ bb,
                                   const int* __restrict__ ei,
                                   _Float16* __restrict__ ea_out,
                                   float* __restrict__ loopacc) {
  __shared__ float tile[4][16][HID];
  int wv = threadIdx.x >> 5;
  int wave = blockIdx.x * 4 + wv;     // exact: E/16 waves, divisible by 4
  int e0 = wave * 16;
  int lane = threadIdx.x & 31;
  v8f acc[8] = {};
  v16h a = load_a_f32(ea_in, EDGE_IN, e0, 0, lane);
#pragma unroll
  for (int j = 0; j < 8; ++j) {
    v16h b = load_b_packed(Bw, 8, 0, j, lane);
    acc[j] = wmma_fma(a, b, acc[j]);
  }
  int mb = (lane >> 4) * 8, nloc = lane & 15;
#pragma unroll
  for (int j = 0; j < 8; ++j) {
    int c = j * 16 + nloc;
    float bv = eb[c];
#pragma unroll
    for (int v = 0; v < 8; ++v) tile[wv][mb + v][c] = acc[j][v] + bv;
  }
  __syncthreads();
  for (int r = 0; r < 16; ++r) {
    float vv[4];
    float s = 0.0f;
#pragma unroll
    for (int i = 0; i < 4; ++i) { vv[i] = tile[wv][r][lane + 32 * i]; s += vv[i]; }
#pragma unroll
    for (int off = 16; off > 0; off >>= 1) s += __shfl_xor(s, off, 32);
    float mean = s * (1.0f / HID);
    float q = 0.0f;
#pragma unroll
    for (int i = 0; i < 4; ++i) { float d = vv[i] - mean; q += d * d; }
#pragma unroll
    for (int off = 16; off > 0; off >>= 1) q += __shfl_xor(q, off, 32);
    float rstd = rsqrtf(q * (1.0f / HID) + 1e-5f);
    int e = e0 + r;
    int dst = ei[N_EDGES + e];
#pragma unroll
    for (int i = 0; i < 4; ++i) {
      int c = lane + 32 * i;
      float y = gelu_exact((vv[i] - mean) * rstd * g[c] + bb[c]);
      ea_out[(size_t)e * HID + c] = (_Float16)y;
      atomicAdd(&loopacc[(size_t)dst * HID + c], y);
    }
  }
}

__global__ void cnt_kernel(const int* __restrict__ ei, float* __restrict__ cnt) {
  int e = blockIdx.x * blockDim.x + threadIdx.x;
  if (e >= N_EDGES) return;
  atomicAdd(&cnt[ei[N_EDGES + e]], 1.0f);
}

__global__ void loop_attr_kernel(const float* __restrict__ loopacc, const float* __restrict__ cnt,
                                 _Float16* __restrict__ ea_out) {
  int tid = blockIdx.x * blockDim.x + threadIdx.x;
  if (tid >= N_NODES * HID) return;
  int n = tid >> 7;
  float d = fmaxf(cnt[n], 1.0f);
  ea_out[(size_t)(N_EDGES + n) * HID + (tid & 127)] = (_Float16)(loopacc[tid] / d);
}

// ---- fused edge GEMM (em = ea_full @ cWe) + GATv2 logit + atomic seg-max ----
__global__ void edge_logit_kernel(const _Float16* __restrict__ ea,
                                  const _Float16* __restrict__ Bw,
                                  const float* __restrict__ xl, const float* __restrict__ xr,
                                  const float* __restrict__ catt,
                                  const int* __restrict__ ei,
                                  float* __restrict__ logits, unsigned* __restrict__ segmax) {
  int wave = blockIdx.x * (blockDim.x >> 5) + (threadIdx.x >> 5);
  int e0 = wave * 16;
  if (e0 >= EF) return;
  int lane = threadIdx.x & 31;
  v8f acc[8] = {};
#pragma unroll
  for (int kt = 0; kt < 4; ++kt) {
    v16h a = load_a_f16(ea, HID, e0, kt * 32, lane);
#pragma unroll
    for (int j = 0; j < 8; ++j) {
      v16h b = load_b_packed(Bw, 8, kt, j, lane);
      acc[j] = wmma_fma(a, b, acc[j]);
    }
  }
  int mb = (lane >> 4) * 8, nloc = lane & 15;
  int se[8], de[8];
#pragma unroll
  for (int v = 0; v < 8; ++v) {
    int e = e0 + mb + v;
    if (e < N_EDGES) { se[v] = ei[e]; de[v] = ei[N_EDGES + e]; }
    else { se[v] = e - N_EDGES; de[v] = e - N_EDGES; }
  }
#pragma unroll
  for (int j = 0; j < 8; ++j) {
    int c = j * 16 + nloc;
    float av = catt[c];
#pragma unroll
    for (int v = 0; v < 8; ++v) {
      float m = acc[j][v] + xl[(size_t)se[v] * HID + c] + xr[(size_t)de[v] * HID + c];
      m = (m > 0.0f) ? m : 0.2f * m;
      float t = m * av;
      t += __shfl_xor(t, 1, 32);
      t += __shfl_xor(t, 2, 32);
      t += __shfl_xor(t, 4, 32);
      t += __shfl_xor(t, 8, 32);   // sum over 16-lane group (head dim)
      if (nloc == 0) {
        int e = e0 + mb + v;
        logits[(size_t)e * HEADS + j] = t;
        atomicMax(&segmax[(size_t)de[v] * HEADS + j], fkey(t));
      }
    }
  }
}

__global__ void softmax_ex_kernel(const int* __restrict__ ei, const unsigned* __restrict__ segmax,
                                  float* __restrict__ logits, float* __restrict__ den) {
  int tid = blockIdx.x * blockDim.x + threadIdx.x;
  if (tid >= EF * HEADS) return;
  int e = tid >> 3, j = tid & 7;
  int d = (e < N_EDGES) ? ei[N_EDGES + e] : (e - N_EDGES);
  float mx = funkey(segmax[d * HEADS + j]);
  float ex = expf(logits[tid] - mx);
  logits[tid] = ex;
  atomicAdd(&den[d * HEADS + j], ex);
}

__global__ void scatter_out_kernel(const int* __restrict__ ei, const float* __restrict__ exbuf,
                                   const float* __restrict__ den, const float* __restrict__ xl,
                                   float* __restrict__ outacc) {
  long tid = (long)blockIdx.x * blockDim.x + threadIdx.x;
  if (tid >= (long)EF * HID) return;
  int e = (int)(tid >> 7), c = (int)(tid & 127), j = c >> 4;
  int s, d;
  if (e < N_EDGES) { s = ei[e]; d = ei[N_EDGES + e]; } else { s = d = e - N_EDGES; }
  float alpha = exbuf[(size_t)e * HEADS + j] / (den[d * HEADS + j] + 1e-16f);
  atomicAdd(&outacc[(size_t)d * HID + c], alpha * xl[(size_t)s * HID + c]);
}

__global__ void update_h_kernel(const float* __restrict__ outacc, const float* __restrict__ cbias,
                                float* __restrict__ h, float* __restrict__ hidden) {
  int tid = blockIdx.x * blockDim.x + threadIdx.x;
  if (tid >= N_NODES * HID) return;
  int c = tid & (HID - 1);
  float y = gelu_exact(outacc[tid] + cbias[c]) + hidden[tid];
  h[tid] = y;
  hidden[tid] = y;
}

__global__ void pool_accum_kernel(const float* __restrict__ h, const int* __restrict__ batch,
                                  float* __restrict__ gfeat, float* __restrict__ gcnt) {
  int tid = blockIdx.x * blockDim.x + threadIdx.x;
  if (tid >= N_NODES * HID) return;
  int n = tid >> 7, c = tid & 127;
  int b = batch[n];
  atomicAdd(&gfeat[b * HID + c], h[tid]);
  if (c == 0) atomicAdd(&gcnt[b], 1.0f);
}

__global__ void pool_div_kernel(float* __restrict__ gfeat, const float* __restrict__ gcnt) {
  int tid = blockIdx.x * blockDim.x + threadIdx.x;
  if (tid >= N_GRAPHS * HID) return;
  gfeat[tid] /= fmaxf(gcnt[tid >> 7], 1.0f);
}

// ------------------------------- host side -----------------------------------
extern "C" void kernel_launch(void* const* d_in, const int* in_sizes, int n_in,
                              void* d_out, int out_size, void* d_ws, size_t ws_size,
                              hipStream_t stream) {
  (void)in_sizes; (void)n_in; (void)out_size; (void)ws_size;
  const float* x     = (const float*)d_in[0];
  const float* eattr = (const float*)d_in[1];
  const float* t     = (const float*)d_in[2];
  const float* cond  = (const float*)d_in[3];
  const float* tW1   = (const float*)d_in[4];
  const float* tb1   = (const float*)d_in[5];
  const float* tW2   = (const float*)d_in[6];
  const float* tb2   = (const float*)d_in[7];
  const float* encW  = (const float*)d_in[8];
  const float* encb  = (const float*)d_in[9];
  const float* encg  = (const float*)d_in[10];
  const float* encbe = (const float*)d_in[11];
  const float* eW    = (const float*)d_in[12];
  const float* eb    = (const float*)d_in[13];
  const float* eg    = (const float*)d_in[14];
  const float* ebe   = (const float*)d_in[15];
  const float* cWl   = (const float*)d_in[16];
  const float* cbl   = (const float*)d_in[17];
  const float* cWr   = (const float*)d_in[18];
  const float* cbr   = (const float*)d_in[19];
  const float* cWe   = (const float*)d_in[20];
  const float* catt  = (const float*)d_in[21];
  const float* cbias = (const float*)d_in[22];
  const float* nW1   = (const float*)d_in[23];
  const float* nb1   = (const float*)d_in[24];
  const float* ng    = (const float*)d_in[25];
  const float* nbe   = (const float*)d_in[26];
  const float* nW2   = (const float*)d_in[27];
  const float* nb2   = (const float*)d_in[28];
  const float* pW1   = (const float*)d_in[29];
  const float* pb1   = (const float*)d_in[30];
  const float* pg    = (const float*)d_in[31];
  const float* pbe   = (const float*)d_in[32];
  const float* pW2   = (const float*)d_in[33];
  const float* pb2   = (const float*)d_in[34];
  const int*   ei    = (const int*)d_in[35];
  const int*   batch = (const int*)d_in[36];
  float* outp = (float*)d_out;

  char* ws = (char*)d_ws;
  size_t off = 0;
  auto alloc = [&](size_t bytes) -> void* {
    off = (off + 255) & ~(size_t)255;
    void* p = ws + off;
    off += bytes;
    return p;
  };

  _Float16* wEnc = (_Float16*)alloc(96 * 128 * 2);
  _Float16* wE   = (_Float16*)alloc(32 * 128 * 2);
  _Float16* wCl  = (_Float16*)alloc(512 * 128 * 2);
  _Float16* wCr  = (_Float16*)alloc(512 * 128 * 2);
  _Float16* wCe  = (_Float16*)alloc(512 * 128 * 2);
  _Float16* wN1  = (_Float16*)alloc(128 * 128 * 2);
  _Float16* wN2  = (_Float16*)alloc(128 * 64 * 2);
  _Float16* wP1  = (_Float16*)alloc(128 * 128 * 2);
  _Float16* wP2  = (_Float16*)alloc(128 * 16 * 2);
  float* temb    = (float*)alloc(16 * 128 * 4);
  float* Acat    = (float*)alloc((size_t)N_NODES * 96 * 4);
  float* pre     = (float*)alloc((size_t)N_NODES * HID * 4);
  float* h       = (float*)alloc((size_t)N_NODES * HID * 4);
  float* hidden  = (float*)alloc((size_t)N_NODES * HID * 4);
  float* xlbuf   = (float*)alloc((size_t)N_NODES * HID * 4);
  float* xrbuf   = (float*)alloc((size_t)N_NODES * HID * 4);
  float* outacc  = (float*)alloc((size_t)N_NODES * HID * 4);
  float* loopacc = (float*)alloc((size_t)N_NODES * HID * 4);
  float* cnt     = (float*)alloc((size_t)N_NODES * 4);
  _Float16* eaF  = (_Float16*)alloc((size_t)EF * HID * 2);
  float* logits  = (float*)alloc((size_t)EF * HEADS * 4);
  unsigned* smax = (unsigned*)alloc((size_t)N_NODES * HEADS * 4);
  float* sden    = (float*)alloc((size_t)N_NODES * HEADS * 4);
  float* gfeat   = (float*)alloc(16 * 128 * 4);
  float* gcnt    = (float*)alloc(16 * 4);
  float* ptmp    = (float*)alloc(16 * 128 * 4);
  float* ptmp2   = (float*)alloc(16 * 128 * 4);

  auto cdiv = [](long a, long b) { return (int)((a + b - 1) / b); };

  // 1) weights -> f16 packed WMMA-B fragments (pack each matrix separately!)
  cvt_pack_kernel<<<cdiv(96 * 128, 256), 256, 0, stream>>>(encW, 72, 128, wEnc, 96, 128);
  cvt_pack_kernel<<<cdiv(32 * 128, 256), 256, 0, stream>>>(eW, 32, 128, wE, 32, 128);
  for (int l = 0; l < N_LAYERS; ++l) {
    cvt_pack_kernel<<<cdiv(128 * 128, 256), 256, 0, stream>>>(cWl + (size_t)l * 16384, 128, 128,
                                                              wCl + (size_t)l * 16384, 128, 128);
    cvt_pack_kernel<<<cdiv(128 * 128, 256), 256, 0, stream>>>(cWr + (size_t)l * 16384, 128, 128,
                                                              wCr + (size_t)l * 16384, 128, 128);
    cvt_pack_kernel<<<cdiv(128 * 128, 256), 256, 0, stream>>>(cWe + (size_t)l * 16384, 128, 128,
                                                              wCe + (size_t)l * 16384, 128, 128);
  }
  cvt_pack_kernel<<<cdiv(128 * 128, 256), 256, 0, stream>>>(nW1, 128, 128, wN1, 128, 128);
  cvt_pack_kernel<<<cdiv(128 * 64, 256), 256, 0, stream>>>(nW2, 128, 64, wN2, 128, 64);
  cvt_pack_kernel<<<cdiv(128 * 128, 256), 256, 0, stream>>>(pW1, 128, 128, wP1, 128, 128);
  cvt_pack_kernel<<<cdiv(128 * 16, 256), 256, 0, stream>>>(pW2, 128, 8, wP2, 128, 16);

  // 2) time MLP
  time_mlp_kernel<<<N_GRAPHS, 128, 0, stream>>>(t, tW1, tb1, tW2, tb2, temb);

  // 3) node encoder: concat -> GEMM -> LN+GELU + t_emb[batch]; hidden = h
  concat_kernel<<<cdiv((long)N_NODES * 96, 256), 256, 0, stream>>>(x, cond, batch, Acat);
  {
    int waves = N_NODES / 16;
    gemm_wmma_kernel<8><<<cdiv(waves, 4), 128, 0, stream>>>(Acat, N_NODES, 96, wEnc, encb, pre, 128);
  }
  ln_gelu_kernel<<<cdiv(N_NODES, 8), 256, 0, stream>>>(pre, encg, encbe, h, hidden, temb, batch, N_NODES);

  // 4) edge encoder (fused GEMM+LN+GELU) + self-loop mean attrs
  zero_kernel<<<cdiv((long)N_NODES * HID, 256), 256, 0, stream>>>(loopacc, (long)N_NODES * HID);
  zero_kernel<<<cdiv(N_NODES, 256), 256, 0, stream>>>(cnt, N_NODES);
  edge_encode_kernel<<<(N_EDGES / 16) / 4, 128, 0, stream>>>(eattr, wE, eb, eg, ebe, ei, eaF, loopacc);
  cnt_kernel<<<cdiv(N_EDGES, 256), 256, 0, stream>>>(ei, cnt);
  loop_attr_kernel<<<cdiv((long)N_NODES * HID, 256), 256, 0, stream>>>(loopacc, cnt, eaF);

  // 5) GATv2 layers
  for (int l = 0; l < N_LAYERS; ++l) {
    const _Float16* Bl = wCl + (size_t)l * 128 * 128;
    const _Float16* Br = wCr + (size_t)l * 128 * 128;
    const _Float16* Be = wCe + (size_t)l * 128 * 128;
    int waves = N_NODES / 16;
    gemm_wmma_kernel<8><<<cdiv(waves, 4), 128, 0, stream>>>(h, N_NODES, 128, Bl, cbl + l * 128, xlbuf, 128);
    gemm_wmma_kernel<8><<<cdiv(waves, 4), 128, 0, stream>>>(h, N_NODES, 128, Br, cbr + l * 128, xrbuf, 128);
    zero_kernel<<<cdiv((long)N_NODES * HEADS, 256), 256, 0, stream>>>((float*)smax, (long)N_NODES * HEADS);
    zero_kernel<<<cdiv((long)N_NODES * HEADS, 256), 256, 0, stream>>>(sden, (long)N_NODES * HEADS);
    zero_kernel<<<cdiv((long)N_NODES * HID, 256), 256, 0, stream>>>(outacc, (long)N_NODES * HID);
    int ewaves = EF / 16;
    edge_logit_kernel<<<cdiv(ewaves, 4), 128, 0, stream>>>(eaF, Be, xlbuf, xrbuf, catt + l * 128, ei, logits, smax);
    softmax_ex_kernel<<<cdiv((long)EF * HEADS, 256), 256, 0, stream>>>(ei, smax, logits, sden);
    scatter_out_kernel<<<cdiv((long)EF * HID, 256), 256, 0, stream>>>(ei, logits, sden, xlbuf, outacc);
    update_h_kernel<<<cdiv((long)N_NODES * HID, 256), 256, 0, stream>>>(outacc, cbias + l * 128, h, hidden);
  }

  // 6) pred_noise head
  {
    int waves = N_NODES / 16;
    gemm_wmma_kernel<8><<<cdiv(waves, 4), 128, 0, stream>>>(h, N_NODES, 128, wN1, nb1, pre, 128);
    ln_gelu_kernel<<<cdiv(N_NODES, 8), 256, 0, stream>>>(pre, ng, nbe, xlbuf, nullptr, nullptr, nullptr, N_NODES);
    gemm_wmma_kernel<4><<<cdiv(waves, 4), 128, 0, stream>>>(xlbuf, N_NODES, 128, wN2, nb2, outp, 64);
  }

  // 7) pooling + pred_props head
  zero_kernel<<<cdiv(16 * 128, 256), 256, 0, stream>>>(gfeat, 16 * 128);
  zero_kernel<<<1, 256, 0, stream>>>(gcnt, 16);
  pool_accum_kernel<<<cdiv((long)N_NODES * HID, 256), 256, 0, stream>>>(h, batch, gfeat, gcnt);
  pool_div_kernel<<<cdiv(16 * 128, 256), 256, 0, stream>>>(gfeat, gcnt);
  gemm_wmma_kernel<8><<<1, 128, 0, stream>>>(gfeat, N_GRAPHS, 128, wP1, pb1, ptmp, 128);
  ln_gelu_kernel<<<1, 256, 0, stream>>>(ptmp, pg, pbe, ptmp2, nullptr, nullptr, nullptr, N_GRAPHS);
  gemm_wmma_kernel<1><<<1, 128, 0, stream>>>(ptmp2, N_GRAPHS, 128, wP2, pb2,
                                             outp + (size_t)N_NODES * NODE_OUT, COND);
}
